// Model_57363583205453
// MI455X (gfx1250) — matrix operand order repaired and checked by reading, not compiled, not run on hardware
//
#include <hip/hip_runtime.h>
#include <math.h>

typedef __attribute__((ext_vector_type(16))) _Float16 v16h;
typedef __attribute__((ext_vector_type(8)))  _Float16 v8h;
typedef __attribute__((ext_vector_type(8)))  float    v8f;

#define NROWS 131072   // R*S = 4096*32
#define MSAMP 6

// Level metadata (compile-time, matches _level_meta())
__constant__ int      c_off[10] = {0,4920,40864,315496,2412648,4509800,6606952,8704104,10801256,12898408};
__constant__ unsigned c_res[3]  = {17u,33u,65u};
__constant__ unsigned c_siz[3]  = {4920u,35944u,274632u};

// ---------------- hash-grid encode: one thread per (r,s) row ----------------
__global__ __launch_bounds__(256) void encode_kernel(
    const float* __restrict__ means, const float* __restrict__ stds,
    const float* __restrict__ table, _Float16* __restrict__ feat16)
{
  int row = blockIdx.x * blockDim.x + threadIdx.x;
  if (row >= NROWS) return;
  float acc[40];
#pragma unroll
  for (int i = 0; i < 40; ++i) acc[i] = 0.f;

  for (int m = 0; m < MSAMP; ++m) {
    size_t pb = ((size_t)row * MSAMP + m) * 3;
    float x0 = (means[pb + 0] + 1.f) * 0.5f;
    float x1 = (means[pb + 1] + 1.f) * 0.5f;
    float x2 = (means[pb + 2] + 1.f) * 0.5f;
    float sd = stds[(size_t)row * MSAMP + m];
#pragma unroll
    for (int l = 0; l < 10; ++l) {
      float scale = (float)(16 << l);
      float p0 = x0 * scale + 0.5f, p1 = x1 * scale + 0.5f, p2 = x2 * scale + 0.5f;
      float f0 = floorf(p0), f1 = floorf(p1), f2 = floorf(p2);
      float r0 = p0 - f0, r1 = p1 - f1, r2 = p2 - f2;
      unsigned u0 = (unsigned)f0, u1 = (unsigned)f1, u2 = (unsigned)f2;
      float gw = erff(1.0f / (2.8284271247461903f * sd * scale));
      float a0 = 0.f, a1 = 0.f, a2 = 0.f, a3 = 0.f;
#pragma unroll
      for (int c = 0; c < 8; ++c) {
        unsigned b0 = c & 1u, b1 = (c >> 1) & 1u, b2 = (c >> 2) & 1u;
        unsigned cx = u0 + b0, cy = u1 + b1, cz = u2 + b2;
        unsigned idx;
        if (l < 3) {
          unsigned R = c_res[l];
          idx = (cx * R + cy) * R + cz;
          unsigned mx = c_siz[l] - 1u;           // match JAX OOB-gather clamp
          idx = (idx > mx) ? mx : idx;
        } else {
          idx = (cx * 1u) ^ (cy * 2654435761u) ^ (cz * 805459861u);
          idx &= 2097151u;                        // HASHMAP_SIZE - 1
        }
        float w = (b0 ? r0 : 1.f - r0) * (b1 ? r1 : 1.f - r1) * (b2 ? r2 : 1.f - r2);
        const float4 t = *(const float4*)(table + ((size_t)c_off[l] + idx) * 4);
        a0 += w * t.x; a1 += w * t.y; a2 += w * t.z; a3 += w * t.w;
      }
      acc[l * 4 + 0] += gw * a0;
      acc[l * 4 + 1] += gw * a1;
      acc[l * 4 + 2] += gw * a2;
      acc[l * 4 + 3] += gw * a3;
    }
  }
  _Float16* o = feat16 + (size_t)row * 64;
  const float inv = 1.f / 6.f;
#pragma unroll
  for (int i = 0; i < 40; ++i) o[i] = (_Float16)(acc[i] * inv);
#pragma unroll
  for (int i = 40; i < 64; ++i) o[i] = (_Float16)0.f;
}

// ---------------- view-dir positional encoding into inp16 cols 256..287 ----
__global__ __launch_bounds__(256) void direnc_kernel(
    const float* __restrict__ vd, _Float16* __restrict__ inp16)
{
  int row = blockIdx.x * blockDim.x + threadIdx.x;
  if (row >= NROWS) return;
  int r = row >> 5;  // S = 32
  float d[3] = { vd[r * 3 + 0], vd[r * 3 + 1], vd[r * 3 + 2] };
  _Float16* o = inp16 + (size_t)row * 288 + 256;
#pragma unroll
  for (int j = 0; j < 3; ++j) o[j] = (_Float16)d[j];
#pragma unroll
  for (int sdeg = 0; sdeg < 4; ++sdeg) {
    float sc = (float)(1 << sdeg);
#pragma unroll
    for (int j = 0; j < 3; ++j) {
      float xb = d[j] * sc;
      o[3  + sdeg * 3 + j] = (_Float16)sinf(xb);   // sin(xb)
      o[15 + sdeg * 3 + j] = (_Float16)cosf(xb);   // sin(xb + pi/2)
    }
  }
#pragma unroll
  for (int j = 27; j < 32; ++j) o[j] = (_Float16)0.f;  // K padding 283..287
}

// ---------------- weight convert: W[K][N] f32 -> WT[N][Kpad] f16 ------------
__global__ __launch_bounds__(256) void wconv_kernel(
    const float* __restrict__ W, _Float16* __restrict__ WT, int K, int N, int Kpad)
{
  int t = blockIdx.x * blockDim.x + threadIdx.x;
  if (t >= N * Kpad) return;
  int n = t / Kpad, k = t % Kpad;
  float v = (k < K) ? W[(size_t)k * N + n] : 0.f;
  WT[t] = (_Float16)v;
}

// ---------------- generic WMMA GEMM: out = act(A @ W + b) -------------------
// Compile-time KPAD / KSPLIT: inner loop fully unrolls with immediate offsets.
// A rows: k <  KSPLIT  -> A0[row][k]        (lda0)
//         k >= KSPLIT  -> A1[row][k-KSPLIT] (lda1)
// WT: [Ncols][KPAD] f16. 8 waves/block; wave tile 16(M)x32(N) (two 16x16
// accumulators sharing one A fragment). Block tile 128 rows x 32 cols.
// grid = (NROWS/128, Ncols/32).
template<int KPAD, int KSPLIT>
__global__ __launch_bounds__(256) void gemm_wmma(
    const _Float16* __restrict__ A0, int lda0,
    const _Float16* __restrict__ A1, int lda1,
    const _Float16* __restrict__ WT,
    const float* __restrict__ bias,
    _Float16* __restrict__ out16, int ldo,
    int relu,
    float* __restrict__ density)     // non-null: layer-2 fused softplus on col 0
{
  int wave = threadIdx.x >> 5;
  int lane = threadIdx.x & 31;
  int rowBase = blockIdx.x * 128 + wave * 16;
  int colBase = blockIdx.y * 32;
  int kh  = lane >> 4;     // lane half
  int l16 = lane & 15;

  // A fragment (16-bit A 16x32 layout): per k-step, two contiguous 8-half
  // chunks at k = kb + kh*8 and k = kb + 16 + kh*8.
  const _Float16* Arow0 = A0 + (size_t)(rowBase + l16) * lda0 + kh * 8;
  const _Float16* Arow1 = (KSPLIT < KPAD)
      ? (A1 + (size_t)(rowBase + l16) * lda1 + kh * 8) : nullptr;
  // B fragment: same as A, chunks at k = kb + kh*8 and k = kb + 16 + kh*8.
  const _Float16* Brow0 = WT + (size_t)(colBase + l16) * KPAD + kh * 8;
  const _Float16* Brow1 = WT + (size_t)(colBase + 16 + l16) * KPAD + kh * 8;

  v8f acc0 = {}, acc1 = {};
#pragma unroll
  for (int kb = 0; kb < KPAD; kb += 32) {
    const _Float16* ap = (kb < KSPLIT) ? (Arow0 + kb) : (Arow1 + (kb - KSPLIT));
    v8h a0  = *(const v8h*)(ap);
    v8h a1  = *(const v8h*)(ap + 16);
    v8h b00 = *(const v8h*)(Brow0 + kb);
    v8h b01 = *(const v8h*)(Brow0 + kb + 16);
    v8h b10 = *(const v8h*)(Brow1 + kb);
    v8h b11 = *(const v8h*)(Brow1 + kb + 16);
    v16h a, b0, b1;
#pragma unroll
    for (int i = 0; i < 8; ++i) {
      a[i]  = a0[i];  a[i + 8]  = a1[i];
      b0[i] = b00[i]; b0[i + 8] = b01[i];
      b1[i] = b10[i]; b1[i + 8] = b11[i];
    }
    acc0 = __builtin_amdgcn_wmma_f32_16x16x32_f16(false, a, false, b0, (short)0, acc0, false, false);
    acc1 = __builtin_amdgcn_wmma_f32_16x16x32_f16(false, a, false, b1, (short)0, acc1, false, false);
  }

  int n0 = colBase + l16;
  int n1 = n0 + 16;
  float bv0 = bias[n0];
  float bv1 = bias[n1];
#pragma unroll
  for (int v = 0; v < 8; ++v) {
    int m = rowBase + v + 8 * kh;                 // C/D layout: M = v + 8*(lane/16)
    float val0 = acc0[v] + bv0;
    float val1 = acc1[v] + bv1;
    if (relu) { val0 = fmaxf(val0, 0.f); val1 = fmaxf(val1, 0.f); }
    _Float16* orow = out16 + (size_t)m * ldo;
    orow[n0] = (_Float16)val0;
    orow[n1] = (_Float16)val1;
    if (density && n0 == 0) {
      float t = val0 - 1.0f;                       // softplus(x - 1), stable
      density[m] = fmaxf(t, 0.f) + log1pf(expf(-fabsf(t)));
    }
  }
}

// ---------------- rgb head: sigmoid(h2 @ rw + rb) scaled --------------------
__global__ __launch_bounds__(256) void rgb_kernel(
    const _Float16* __restrict__ h2, const float* __restrict__ rw,
    const float* __restrict__ rb, float* __restrict__ out)
{
  int row = blockIdx.x * blockDim.x + threadIdx.x;
  if (row >= NROWS) return;
  const _Float16* h = h2 + (size_t)row * 256;
  float s0 = 0.f, s1 = 0.f, s2 = 0.f;
#pragma unroll 8
  for (int k = 0; k < 256; ++k) {
    float v = (float)h[k];
    s0 += v * rw[k * 3 + 0];
    s1 += v * rw[k * 3 + 1];
    s2 += v * rw[k * 3 + 2];
  }
  float ss[3] = { s0 + rb[0], s1 + rb[1], s2 + rb[2] };
#pragma unroll
  for (int c = 0; c < 3; ++c) {
    float sg = 1.f / (1.f + expf(-ss[c]));
    out[(size_t)row * 3 + c] = sg * 1.002f - 0.001f;
  }
}

extern "C" void kernel_launch(void* const* d_in, const int* in_sizes, int n_in,
                              void* d_out, int out_size, void* d_ws, size_t ws_size,
                              hipStream_t stream)
{
  const float* means = (const float*)d_in[0];
  const float* stds  = (const float*)d_in[1];
  const float* vdirs = (const float*)d_in[2];
  const float* table = (const float*)d_in[3];
  const float* dw1   = (const float*)d_in[4];
  const float* db1   = (const float*)d_in[5];
  const float* dw2   = (const float*)d_in[6];
  const float* db2   = (const float*)d_in[7];
  const float* vw0   = (const float*)d_in[8];
  const float* vb0   = (const float*)d_in[9];
  const float* vw1   = (const float*)d_in[10];
  const float* vb1   = (const float*)d_in[11];
  const float* rw    = (const float*)d_in[12];
  const float* rb    = (const float*)d_in[13];

  float* out      = (float*)d_out;
  float* rgb_out  = out;                         // (R,S,3) flat
  float* dens_out = out + (size_t)NROWS * 3;     // (R,S) flat

  char* ws = (char*)d_ws;
  size_t off = 0;
  auto wsalloc = [&](size_t bytes) -> void* {
    void* p = ws + off; off += (bytes + 255) & ~(size_t)255; return p;
  };
  _Float16* inp16 = (_Float16*)wsalloc((size_t)NROWS * 288 * 2);  // x | dir | pad
  _Float16* h2a16 = (_Float16*)wsalloc((size_t)NROWS * 256 * 2);
  _Float16* dw1T  = (_Float16*)wsalloc((size_t)64  * 64  * 2);
  _Float16* dw2T  = (_Float16*)wsalloc((size_t)256 * 64  * 2);
  _Float16* vw0T  = (_Float16*)wsalloc((size_t)256 * 288 * 2);
  _Float16* vw1T  = (_Float16*)wsalloc((size_t)256 * 544 * 2);
  // Union region: feat16[0..16MB) + h1_16[16..32MB) early; h2_16 (64MB) reuses
  // the whole region after both are dead (post-GEMM2).
  char* uni = (char*)wsalloc((size_t)NROWS * 256 * 2);
  _Float16* feat16 = (_Float16*)uni;
  _Float16* h1_16  = (_Float16*)(uni + (size_t)NROWS * 64 * 2);
  _Float16* h2_16  = (_Float16*)uni;

  dim3 blk(256);

  wconv_kernel<<<dim3((64  * 64  + 255) / 256), blk, 0, stream>>>(dw1, dw1T, 40,  64,  64);
  wconv_kernel<<<dim3((256 * 64  + 255) / 256), blk, 0, stream>>>(dw2, dw2T, 64,  256, 64);
  wconv_kernel<<<dim3((256 * 288 + 255) / 256), blk, 0, stream>>>(vw0, vw0T, 283, 256, 288);
  wconv_kernel<<<dim3((256 * 544 + 255) / 256), blk, 0, stream>>>(vw1, vw1T, 539, 256, 544);

  encode_kernel<<<dim3(NROWS / 256), blk, 0, stream>>>(means, stds, table, feat16);
  direnc_kernel<<<dim3(NROWS / 256), blk, 0, stream>>>(vdirs, inp16);

  // L1: feat(40p64) -> 64, relu
  gemm_wmma<64, 64><<<dim3(NROWS / 128, 64 / 32), blk, 0, stream>>>(
      feat16, 64, nullptr, 0, dw1T, db1, h1_16, 64, 1, nullptr);
  // L2: h1(64) -> 256, no act; x into inp16[:,0:256]; fused density = softplus(x0-1)
  gemm_wmma<64, 64><<<dim3(NROWS / 128, 256 / 32), blk, 0, stream>>>(
      h1_16, 64, nullptr, 0, dw2T, db2, inp16, 288, 0, dens_out);
  // L3: inp(283p288) -> 256, relu
  gemm_wmma<288, 288><<<dim3(NROWS / 128, 256 / 32), blk, 0, stream>>>(
      inp16, 288, nullptr, 0, vw0T, vb0, h2a16, 256, 1, nullptr);
  // L4: concat(h2a(256), inp(283)) = 539p544 -> 256, relu (split-K A source)
  gemm_wmma<544, 256><<<dim3(NROWS / 128, 256 / 32), blk, 0, stream>>>(
      h2a16, 256, inp16, 288, vw1T, vb1, h2_16, 256, 1, nullptr);

  rgb_kernel<<<dim3(NROWS / 256), blk, 0, stream>>>(h2_16, rw, rb, rgb_out);
}
